// MultiLayerGRU_65841848648480
// MI455X (gfx1250) — matrix operand order repaired and checked by reading, not compiled, not run on hardware
//
#include <hip/hip_runtime.h>
#include <hip/hip_bf16.h>
#include <cstdint>

// Problem constants (match reference)
#define B_    64
#define T_    512
#define DIN_  256
#define H_    1024
#define H3_   3072
#define DOUT_ 256

typedef __bf16 bf16_t;
typedef __attribute__((ext_vector_type(16))) __bf16 v16bf;
typedef __attribute__((ext_vector_type(8)))  __bf16 v8bf;
typedef __attribute__((ext_vector_type(8)))  float  v8f;

union ABFrag { v16bf v; v8bf h[2]; };

// A-matrix fragment (16x32 bf16): lane holds row M=lane&15; the lane's 16
// elements are one contiguous 16-element run at K = 16*half.
__device__ __forceinline__ v16bf load_a_frag(const bf16_t* __restrict__ row, int k, int half) {
  ABFrag f;
  f.h[0] = *(const v8bf*)(row + k + 16 * half);
  f.h[1] = *(const v8bf*)(row + k + 16 * half + 8);
  return f.v;
}

__device__ __forceinline__ float sigmoid_f(float x) { return 1.0f / (1.0f + __expf(-x)); }

// LDS weight slice: [buf][gate][row][k0..63] with rows padded 64->72 elements
// (144 B) to spread the row-strided fragment reads across the 64 LDS banks.
#define WROW_ 72

// Per-thread staged registers for one 64-deep K slice (one 16-B chunk per gate).
struct Staged { v8bf g0, g1, g2; };

// ---------------------------------------------------------------------------
// Fused GRU step for one layer. One workgroup (4 waves = 128 threads) owns one
// 16-wide hidden-column slab and ALL 64 batch rows (wave w -> rows 16w..16w+15).
// Each 64-deep K slice of the 3 gate weight tiles (6 KB) is staged into LDS
// exactly once per workgroup: 384 b128 chunks = 3 per thread, software-
// pipelined through VGPRs (loads for slice i+1 issue before slice i's WMMAs,
// ds_store after), double buffered with one barrier per slice. Each wave then
// builds its B fragments with ds_load_b128. Epilogue fuses the GRU gate update
// and writes h as fp32 (carry) + bf16 (next step's A operand).
// K1_ = input width, AS_ = A row stride (compile-time -> shift-only addressing).
// ---------------------------------------------------------------------------
template <int K1_, int AS_>
__global__ __launch_bounds__(128)
void gru_step_kernel(const bf16_t* __restrict__ A1,
                     const bf16_t* __restrict__ Wi, const bf16_t* __restrict__ Wh,
                     const float* __restrict__ bi, const float* __restrict__ bh,
                     const float* __restrict__ hprev_f, const bf16_t* __restrict__ hprev_bf,
                     float* __restrict__ hout_f, bf16_t* __restrict__ hout_bf) {
  __shared__ __align__(16) bf16_t wtile[2][3][16][WROW_];

  const int tid  = (int)threadIdx.x;
  const int lane = tid & 31;
  const int wave = tid >> 5;            // batch-row tile: rows [16*wave, 16*wave+16)
  const int half = lane >> 4;
  const int lcol = lane & 15;
  const int c0 = (int)blockIdx.x * 16;  // hidden-column slab [c0, c0+16)
  const int m0 = wave * 16;
  // Staging assignment: thread t -> row r = t>>3, 16-B chunk kc = t&7 (covers
  // k0 + 8*kc .. +7), one chunk per gate (gates offset by H_ weight rows).
  const int sr = tid >> 3, skc = tid & 7;

  v8f air = {}, aiz = {}, ain = {};
  v8f ahr = {}, ahz = {}, ahn = {};

  // ---------------- GEMM phase over one weight matrix ----------------
  auto run_gemm = [&](const bf16_t* __restrict__ W, int rowlen_log2_unused,
                      const bf16_t* __restrict__ arow, int K, int rowlen,
                      v8f& ar, v8f& az, v8f& an) {
    const bf16_t* sp = W + (long)(c0 + sr) * rowlen + skc * 8;
    const long goff = (long)H_ * rowlen;

    auto stage_load = [&](int k0) -> Staged {
      Staged s;
      s.g0 = *(const v8bf*)(sp + k0);
      s.g1 = *(const v8bf*)(sp + k0 + goff);
      s.g2 = *(const v8bf*)(sp + k0 + 2 * goff);
      return s;
    };
    auto stage_store = [&](int buf, const Staged& s) {
      *(v8bf*)&wtile[buf][0][sr][skc * 8] = s.g0;
      *(v8bf*)&wtile[buf][1][sr][skc * 8] = s.g1;
      *(v8bf*)&wtile[buf][2][sr][skc * 8] = s.g2;
    };
    auto bfrag = [&](int buf, int g, int ks) -> v16bf {
      ABFrag f;
      const bf16_t* p = &wtile[buf][g][lcol][ks + 16 * half];
      f.h[0] = *(const v8bf*)p;
      f.h[1] = *(const v8bf*)(p + 8);
      return f.v;
    };

    Staged s0 = stage_load(0);
    stage_store(0, s0);
    __syncthreads();
    const int nit = K / 64;
#pragma unroll
    for (int it = 0; it < nit; ++it) {
      const int k0 = it * 64;
      const int cur = it & 1;
      Staged nx;
      if (it + 1 < nit) nx = stage_load(k0 + 64);  // issue early: hides L2 latency
#pragma unroll
      for (int ks = 0; ks < 64; ks += 32) {
        const v16bf a = load_a_frag(arow, k0 + ks, half);
        ar = __builtin_amdgcn_wmma_f32_16x16x32_bf16(false, a, false, bfrag(cur, 0, ks),
                                                     (short)0, ar, false, false);
        az = __builtin_amdgcn_wmma_f32_16x16x32_bf16(false, a, false, bfrag(cur, 1, ks),
                                                     (short)0, az, false, false);
        an = __builtin_amdgcn_wmma_f32_16x16x32_bf16(false, a, false, bfrag(cur, 2, ks),
                                                     (short)0, an, false, false);
      }
      if (it + 1 < nit) stage_store(cur ^ 1, nx);  // waits loadcnt here, post-compute
      __syncthreads();
    }
  };

  // ---- Gi = A1 @ Wi^T ----
  run_gemm(Wi, 0, A1 + (long)(m0 + lcol) * AS_, K1_, K1_, air, aiz, ain);
  // ---- Gh = h_prev @ Wh^T ----
  run_gemm(Wh, 0, hprev_bf + (long)(m0 + lcol) * H_, H_, H_, ahr, ahz, ahn);

  // ---- fused GRU gate epilogue (C/D layout: VGPR v -> row v + 8*half, lane -> col) ----
  const int n = c0 + lcol;
  const float bir = bi[n], biz = bi[n + H_], bin = bi[n + 2 * H_];
  const float bhr = bh[n], bhz = bh[n + H_], bhn = bh[n + 2 * H_];
#pragma unroll
  for (int v = 0; v < 8; ++v) {
    const int m = m0 + v + 8 * half;
    const float r  = sigmoid_f((air[v] + bir) + (ahr[v] + bhr));
    const float z  = sigmoid_f((aiz[v] + biz) + (ahz[v] + bhz));
    const float nn = tanhf((ain[v] + bin) + r * (ahn[v] + bhn));
    const float hp = hprev_f[(long)m * H_ + n];
    const float hv = (1.0f - z) * nn + z * hp;
    hout_f[(long)m * H_ + n]  = hv;
    hout_bf[(long)m * H_ + n] = (bf16_t)hv;
  }
}

// ---------------------------------------------------------------------------
// Output projection: out = h1 @ Wo^T + bo   ([64x1024] x [1024x256] -> fp32)
// Runs once; simple one-wave-per-tile version (weights fetched straight from L2).
// ---------------------------------------------------------------------------
__global__ __launch_bounds__(32)
void proj_kernel(const bf16_t* __restrict__ hbf, const bf16_t* __restrict__ Wo,
                 const float* __restrict__ bo, float* __restrict__ out) {
  const int lane = (int)(threadIdx.x & 31u);
  const int half = lane >> 4;
  const int lcol = lane & 15;
  const int c0 = (int)blockIdx.x * 16;
  const int m0 = (int)blockIdx.y * 16;

  v8f acc = {};
  const bf16_t* arow = hbf + (long)(m0 + lcol) * H_;
  const bf16_t* wrow = Wo + (long)(c0 + lcol) * H_;
  for (int k = 0; k < H_; k += 32) {
    const v16bf a = load_a_frag(arow, k, half);
    ABFrag b;
    b.h[0] = *(const v8bf*)(wrow + k + 16 * half);
    b.h[1] = *(const v8bf*)(wrow + k + 16 * half + 8);
    acc = __builtin_amdgcn_wmma_f32_16x16x32_bf16(false, a, false, b.v,
                                                  (short)0, acc, false, false);
  }
  const float bb = bo[c0 + lcol];
#pragma unroll
  for (int v = 0; v < 8; ++v) {
    const int m = m0 + v + 8 * half;
    out[(long)m * DOUT_ + c0 + lcol] = acc[v] + bb;
  }
}

// fp32 -> bf16 one-time conversion (weights + x)
__global__ void cvt_bf16_kernel(const float* __restrict__ s, bf16_t* __restrict__ d, long n) {
  long i = (long)blockIdx.x * blockDim.x + threadIdx.x;
  if (i < n) d[i] = (bf16_t)s[i];
}

__global__ void zero_u32_kernel(uint32_t* __restrict__ p, long n) {
  long i = (long)blockIdx.x * blockDim.x + threadIdx.x;
  if (i < n) p[i] = 0u;
}

extern "C" void kernel_launch(void* const* d_in, const int* in_sizes, int n_in,
                              void* d_out, int out_size, void* d_ws, size_t ws_size,
                              hipStream_t stream) {
  (void)in_sizes; (void)n_in; (void)out_size; (void)ws_size;

  const float* x   = (const float*)d_in[0];
  const float* Wi0 = (const float*)d_in[1];
  const float* Wh0 = (const float*)d_in[2];
  const float* bi0 = (const float*)d_in[3];
  const float* bh0 = (const float*)d_in[4];
  const float* Wi1 = (const float*)d_in[5];
  const float* Wh1 = (const float*)d_in[6];
  const float* bi1 = (const float*)d_in[7];
  const float* bh1 = (const float*)d_in[8];
  const float* Wo  = (const float*)d_in[9];
  const float* bo  = (const float*)d_in[10];

  // Workspace layout (~40 MB total), 256B aligned chunks.
  char* ws = (char*)d_ws;
  size_t off = 0;
  auto take = [&](size_t bytes) -> char* {
    char* p = ws + off;
    off += (bytes + 255) & ~(size_t)255;
    return p;
  };
  bf16_t* x_bf   = (bf16_t*)take((size_t)B_ * T_ * DIN_ * 2);  // 16 MB
  bf16_t* Wi0_bf = (bf16_t*)take((size_t)H3_ * DIN_ * 2);      // 1.5 MB
  bf16_t* Wh0_bf = (bf16_t*)take((size_t)H3_ * H_ * 2);        // 6 MB
  bf16_t* Wi1_bf = (bf16_t*)take((size_t)H3_ * H_ * 2);        // 6 MB
  bf16_t* Wh1_bf = (bf16_t*)take((size_t)H3_ * H_ * 2);        // 6 MB
  bf16_t* Wo_bf  = (bf16_t*)take((size_t)DOUT_ * H_ * 2);      // 0.5 MB
  // h state: ping-pong buffers, fp32 carry + bf16 operand copies (contiguous
  // region so one zero-init kernel covers all of it).
  float*  h0f  = (float*)take((size_t)2 * B_ * H_ * 4);
  float*  h1f  = (float*)take((size_t)2 * B_ * H_ * 4);
  bf16_t* h0bf = (bf16_t*)take((size_t)2 * B_ * H_ * 2);
  bf16_t* h1bf = (bf16_t*)take((size_t)2 * B_ * H_ * 2);

  auto cvt = [&](const float* s, bf16_t* d, long n) {
    cvt_bf16_kernel<<<dim3((unsigned)((n + 255) / 256)), dim3(256), 0, stream>>>(s, d, n);
  };
  cvt(x,   x_bf,   (long)B_ * T_ * DIN_);
  cvt(Wi0, Wi0_bf, (long)H3_ * DIN_);
  cvt(Wh0, Wh0_bf, (long)H3_ * H_);
  cvt(Wi1, Wi1_bf, (long)H3_ * H_);
  cvt(Wh1, Wh1_bf, (long)H3_ * H_);
  cvt(Wo,  Wo_bf,  (long)DOUT_ * H_);

  // Zero all h state (1.5 MB contiguous: h0f,h1f,h0bf,h1bf).
  {
    const long zwords = ((long)2 * B_ * H_ * 4 * 2 + (long)2 * B_ * H_ * 2 * 2) / 4;
    zero_u32_kernel<<<dim3((unsigned)((zwords + 255) / 256)), dim3(256), 0, stream>>>(
        (uint32_t*)h0f, zwords);
  }

  const long BH = (long)B_ * H_;
  for (int t = 0; t < T_; ++t) {
    const int cur = t & 1, nxt = cur ^ 1;
    // Layer 0: inputs x_t (bf16, A row stride T*DIN across batch), recur on h0.
    gru_step_kernel<DIN_, T_ * DIN_><<<dim3(H_ / 16), dim3(128), 0, stream>>>(
        x_bf + (long)t * DIN_,
        Wi0_bf, Wh0_bf, bi0, bh0,
        h0f + (long)cur * BH, h0bf + (long)cur * BH,
        h0f + (long)nxt * BH, h0bf + (long)nxt * BH);
    // Layer 1: inputs h0_new (just written), recur on h1.
    gru_step_kernel<H_, H_><<<dim3(H_ / 16), dim3(128), 0, stream>>>(
        h0bf + (long)nxt * BH,
        Wi1_bf, Wh1_bf, bi1, bh1,
        h1f + (long)cur * BH, h1bf + (long)cur * BH,
        h1f + (long)nxt * BH, h1bf + (long)nxt * BH);
  }
  // T_=512 (even): final h1 lives in ping-pong buffer 0.
  proj_kernel<<<dim3(DOUT_ / 16, B_ / 16), dim3(32), 0, stream>>>(
      h1bf, Wo_bf, bo, (float*)d_out);
}
